// Encoder_66331474919809
// MI455X (gfx1250) — matrix operand order repaired and checked by reading, not compiled, not run on hardware
//
#include <hip/hip_runtime.h>
#include <hip/hip_bf16.h>
#include <math.h>

// ---------------------------------------------------------------- constants
#define LYR 6
#define EMB 512
#define NH  8
#define HDM 64
#define SEQ 1024
#define LAT 256
#define FFN 2048
#define BAT 4
#define ROWS 32              // query rows per attention block

typedef __attribute__((ext_vector_type(16))) __bf16 v16bf;
typedef __attribute__((ext_vector_type(8)))  float  v8f;
typedef __attribute__((ext_vector_type(4))) unsigned tdm_u32x4;
typedef __attribute__((ext_vector_type(8))) int      tdm_i32x8;
typedef __attribute__((ext_vector_type(4))) int      tdm_i32x4;

union Frag { v16bf v; unsigned u[8]; };

#if defined(__HIP_DEVICE_COMPILE__) && __has_builtin(__builtin_amdgcn_tensor_load_to_lds)
#define USE_TDM 1
#else
#define USE_TDM 0
#endif

// Native bf16 convert (RTNE) -> lets the backend pick v_cvt_*bf16* ops.
__device__ __forceinline__ unsigned short bf_bits(float f) {
  __bf16 h = (__bf16)f;
  return __builtin_bit_cast(unsigned short, h);
}

// ---------------------------------------------------------------- TDM staging
#if USE_TDM
// One 2-D fp32 tile (rows x width, row stride in elems) -> contiguous LDS rows.
__device__ __forceinline__ void tdm_load_2d_f32(const float* gsrc, float* ldst,
                                                int width, int rows, int row_stride) {
  unsigned lds = (unsigned)(unsigned long long)(void*)ldst;   // LDS address: low 32 bits of pointer
  unsigned long long ga = (unsigned long long)(const void*)gsrc;
  unsigned g00 = 1u;                                          // count=1
  unsigned g01 = lds;
  unsigned g02 = (unsigned)(ga & 0xffffffffu);
  unsigned g03 = (unsigned)((ga >> 32) & 0x1ffffffu) | (2u << 30);   // type=2
  unsigned d0 = 2u << 16;                                     // data_size=4B
  unsigned d1 = ((unsigned)width & 0xffffu) << 16;            // tensor_dim0
  unsigned d2 = (((unsigned)rows & 0xffffu) << 16);           // tensor_dim1
  unsigned d3 = ((unsigned)width & 0xffffu) << 16;            // tile_dim0
  unsigned d4 = (unsigned)rows & 0xffffu;                     // tile_dim1
  unsigned d5 = (unsigned)row_stride;                         // dim0 stride
#define RFL(x) __builtin_amdgcn_readfirstlane((int)(x))
  tdm_u32x4 g0 = { (unsigned)RFL(g00), (unsigned)RFL(g01),
                   (unsigned)RFL(g02), (unsigned)RFL(g03) };
  tdm_i32x8 g1 = { RFL(d0), RFL(d1), RFL(d2), RFL(d3), RFL(d4), RFL(d5), 0, 0 };
#undef RFL
  tdm_i32x4 z4 = { 0, 0, 0, 0 };
#if defined(__clang_major__) && __clang_major__ >= 23
  tdm_i32x8 z8 = { 0, 0, 0, 0, 0, 0, 0, 0 };
  __builtin_amdgcn_tensor_load_to_lds(g0, g1, z4, z4, z8, 0);
#else
  __builtin_amdgcn_tensor_load_to_lds(g0, g1, z4, z4, 0);
#endif
}
#endif

// All 256 threads call; each of 8 waves DMAs its row-slice (rows % 8 == 0).
__device__ __forceinline__ void stage_tile_f32(float* dst, const float* src,
                                               int width, int rows, int row_stride,
                                               int tid) {
#if USE_TDM
  int wave = tid >> 5;
  int slice = rows >> 3;
  int r0 = wave * slice;
  tdm_load_2d_f32(src + (long)r0 * row_stride, dst + (long)r0 * width,
                  width, slice, row_stride);
#else
  for (int i = tid; i < width * rows; i += 256)
    dst[i] = src[(long)(i / width) * row_stride + (i % width)];
#endif
}

template <int N>
__device__ __forceinline__ void tdm_wait_n() {
#if USE_TDM
  __builtin_amdgcn_s_wait_tensorcnt(N);
#endif
}

// ---------------------------------------------------------------- LDS converts
// contiguous f32 -> contiguous bf16
__device__ __forceinline__ void convert_copy(unsigned short* dst, const float* src,
                                             int count, int tid) {
  for (int i = tid; i < count; i += 256) dst[i] = bf_bits(src[i]);
}
// src [rows][width] f32 (contig) -> dst [width][rows] bf16 (transpose)
__device__ __forceinline__ void convert_transpose(unsigned short* dst, const float* src,
                                                  int rows, int width, int tid) {
  for (int i = tid; i < rows * width; i += 256) {
    int r = i / width, c = i - r * width;
    dst[c * rows + r] = bf_bits(src[i]);
  }
}
// strided 32x32 f32 block (row stride ld) -> dst [32][32] bf16
__device__ __forceinline__ void convert_a32(unsigned short* dst, const float* src,
                                            int ld, int tid) {
  for (int i = tid; i < 32 * 32; i += 256) {
    int r = i >> 5, c = i & 31;
    dst[i] = bf_bits(src[r * ld + c]);
  }
}

// ---------------------------------------------------------------- fragments (bf16 LDS)
// A fragment (16x32 bf16), row-major tile, packed K-pair dword reads
__device__ __forceinline__ v16bf load_a_bf16(const unsigned short* base, int lda, int lane) {
  int m = lane & 15;
  int khalf = (lane >> 4) & 1;
  Frag f;
#pragma unroll
  for (int j = 0; j < 8; ++j) {
    int kb = ((j < 4) ? 0 : 16) + khalf * 8 + 2 * (j & 3);
    f.u[j] = *(const unsigned*)(base + m * lda + kb);
  }
  return f.v;
}
// B fragment (32x16 bf16) from N-major tile Bt[n][k] (k contiguous)
__device__ __forceinline__ v16bf load_b_nmajor_bf16(const unsigned short* base, int ldk,
                                                    int lane) {
  int n = lane & 15;
  int koff = ((lane >> 4) & 1) * 8;
  Frag f;
#pragma unroll
  for (int j = 0; j < 8; ++j)
    f.u[j] = *(const unsigned*)(base + n * ldk + ((j < 4) ? 0 : 16) + koff + 2 * (j & 3));
  return f.v;
}

__device__ __forceinline__ v8f wmma_bf16(v16bf a, v16bf b, v8f c) {
  return __builtin_amdgcn_wmma_f32_16x16x32_bf16(false, a, false, b, (short)0, c, false, false);
}

// ---------------------------------------------------------------- embed
__global__ void embed_kernel(const float* __restrict__ tok_emb,
                             const float* __restrict__ pos_emb,
                             const int* __restrict__ x,
                             float* __restrict__ h) {
  int bt = blockIdx.x;
  int t = bt & (SEQ - 1);
  long tok = (long)x[bt];
  for (int e = threadIdx.x; e < EMB; e += blockDim.x)
    h[(long)bt * EMB + e] = tok_emb[tok * EMB + e] + pos_emb[(long)t * EMB + e];
}

// ---------------------------------------------------------------- generic GEMM
// C[MxN] = act(A[MxK] @ B[KxN] + bias).  fp32 in/out, bf16 WMMA f32-accum.
// HEADED: B element (k,n) at B[(n>>6)*(K*64) + k*64 + (n&63)]  (wq/wk/wv layout)
template <int ACT, int HEADED>
__global__ void gemm_kernel(const float* __restrict__ A, const float* __restrict__ B,
                            const float* __restrict__ bias, float* __restrict__ C,
                            int M, int N, int K) {
  __shared__ float sAf[2][32 * 32];             // staged A (double buffer)
  __shared__ float sBf[2][32 * 128];            // staged B (double buffer)
  __shared__ unsigned short abuf[32 * 32];      // A bf16 [m][k]
  __shared__ unsigned short btile[128 * 32];    // B bf16 n-major [n][k]
  int tid = threadIdx.x, lane = tid & 31, wave = tid >> 5;
  int m0 = blockIdx.x * 32;
  int n0 = blockIdx.y * 128;
  v8f acc0 = {}, acc1 = {};

  auto issue = [&](int k0, int buf) {
    stage_tile_f32(sAf[buf], A + (long)m0 * K + k0, 32, 32, K, tid);
    if (HEADED) {
      int h0 = n0 >> 6;
      stage_tile_f32(sBf[buf],        B + (long)h0 * (K * 64) + (long)k0 * 64, 64, 32, 64, tid);
      stage_tile_f32(sBf[buf] + 2048, B + (long)(h0 + 1) * (K * 64) + (long)k0 * 64, 64, 32, 64, tid);
    } else {
      stage_tile_f32(sBf[buf], B + (long)k0 * N + n0, 128, 32, N, tid);
    }
  };

  int nk = K >> 5;
  issue(0, 0);
  for (int i = 0; i < nk; ++i) {
    int buf = i & 1;
    if (i + 1 < nk) {
      issue((i + 1) << 5, buf ^ 1);
      if (HEADED) tdm_wait_n<3>(); else tdm_wait_n<2>();
    } else {
      tdm_wait_n<0>();
    }
    __syncthreads();                       // staged f32 tile i ready
    convert_copy(abuf, sAf[buf], 32 * 32, tid);
    if (HEADED) {
      convert_transpose(btile,           sBf[buf],        32, 64, tid);
      convert_transpose(btile + 64 * 32, sBf[buf] + 2048, 32, 64, tid);
    } else {
      convert_transpose(btile, sBf[buf], 32, 128, tid);
    }
    __syncthreads();                       // bf16 tiles ready; stage buf free
    v16bf bf = load_b_nmajor_bf16(btile + (wave * 16) * 32, 32, lane);
    v16bf a0 = load_a_bf16(abuf, 32, lane);
    v16bf a1 = load_a_bf16(abuf + 16 * 32, 32, lane);
    acc0 = wmma_bf16(a0, bf, acc0);
    acc1 = wmma_bf16(a1, bf, acc1);
    __syncthreads();                       // fragments consumed before next convert
  }
  int ncol = n0 + wave * 16 + (lane & 15);
  float bv = bias ? bias[ncol] : 0.f;
#pragma unroll
  for (int r = 0; r < 8; ++r) {
    int rr = ((lane >> 4) ? 8 : 0) + r;
    float x0 = acc0[r] + bv;
    float x1 = acc1[r] + bv;
    if (ACT == 1) { x0 = fmaxf(x0, 0.f); x1 = fmaxf(x1, 0.f); }
    C[(long)(m0 + rr) * N + ncol] = x0;
    C[(long)(m0 + 16 + rr) * N + ncol] = x1;
  }
}

// ---------------------------------------------------------------- fused attention
__global__ void attention_kernel(const float* __restrict__ q, const float* __restrict__ k,
                                 const float* __restrict__ v,
                                 const float* __restrict__ weL,  // [H][1024][256]
                                 const float* __restrict__ wdL,  // [H][256][1024]
                                 const float* __restrict__ beL,  // [H][256]
                                 const float* __restrict__ bdL,  // [H][1024]
                                 const unsigned char* __restrict__ mask,  // [B][T]
                                 float* __restrict__ o,          // [B*T][512]
                                 float* __restrict__ lossSlot) {
  extern __shared__ char smem_raw[];
  float* sbuf    = (float*)smem_raw;                 // [32][1024] raw -> rec -> w
  float* latbuf  = sbuf + ROWS * SEQ;                // [32][256]
  float* red     = latbuf + ROWS * LAT;              // [32][8]
  float* rowstat = red + ROWS * 8;                   // [32]
  float* stage   = rowstat + ROWS;                   // 2 x [128][64] f32 staging
  unsigned short* qtileb = (unsigned short*)(stage + 2 * 8192);  // [32][64] bf16
  unsigned short* btile  = qtileb + ROWS * HDM;      // [128][64] bf16 (n-major)
  unsigned short* abuf   = btile + 128 * HDM;        // [32][32] bf16

  const int tid = threadIdx.x, lane = tid & 31, wave = tid >> 5;
  const int bh = blockIdx.y, b = bh >> 3, hh = bh & 7;
  const int row0 = blockIdx.x * ROWS;
  const float scale = 0.125f;                        // 1/sqrt(64)
  const long qkvb = ((long)b * SEQ) * EMB + hh * HDM;
  float* stg0 = stage;
  float* stg1 = stage + 8192;

  // 0) Q tile -> bf16 once (reused for all 8 s-chunks)
  stage_tile_f32(stg0, q + qkvb + (long)row0 * EMB, HDM, ROWS, EMB, tid);
  tdm_wait_n<0>();
  __syncthreads();
  convert_copy(qtileb, stg0, ROWS * HDM, tid);
  __syncthreads();

  // 1) raw = scale * Q @ K^T   (32 x 1024, in sbuf), double-buffered K tiles
  stage_tile_f32(stg0, k + qkvb, HDM, 128, EMB, tid);
  for (int sc = 0; sc < SEQ / 128; ++sc) {
    float* cur = (sc & 1) ? stg1 : stg0;
    if (sc + 1 < SEQ / 128) {
      stage_tile_f32((sc & 1) ? stg0 : stg1, k + qkvb + (long)(sc + 1) * 128 * EMB,
                     HDM, 128, EMB, tid);
      tdm_wait_n<1>();
    } else {
      tdm_wait_n<0>();
    }
    __syncthreads();
    convert_copy(btile, cur, 128 * HDM, tid);        // [s][d] already n-major
    __syncthreads();
    v8f acc0 = {}, acc1 = {};
#pragma unroll
    for (int kk = 0; kk < HDM; kk += 32) {
      v16bf bf = load_b_nmajor_bf16(btile + (wave * 16) * HDM + kk, HDM, lane);
      v16bf a0 = load_a_bf16(qtileb + kk, HDM, lane);
      v16bf a1 = load_a_bf16(qtileb + 16 * HDM + kk, HDM, lane);
      acc0 = wmma_bf16(a0, bf, acc0);
      acc1 = wmma_bf16(a1, bf, acc1);
    }
    int ncol = sc * 128 + wave * 16 + (lane & 15);
#pragma unroll
    for (int r = 0; r < 8; ++r) {
      int rr = ((lane >> 4) ? 8 : 0) + r;
      sbuf[rr * SEQ + ncol] = acc0[r] * scale;
      sbuf[(16 + rr) * SEQ + ncol] = acc1[r] * scale;
    }
    __syncthreads();
  }

  // 2) lat = relu(raw @ we + be)   (32 x 256, in latbuf)
  for (int nc = 0; nc < LAT / 128; ++nc) {
    int n0 = nc * 128;
    v8f acc0 = {}, acc1 = {};
    stage_tile_f32(stg0, weL + (long)hh * SEQ * LAT + n0, 128, 32, LAT, tid);
    for (int ks = 0; ks < SEQ / 32; ++ks) {
      int k0 = ks * 32;
      float* cur = (ks & 1) ? stg1 : stg0;
      if (ks + 1 < SEQ / 32) {
        stage_tile_f32((ks & 1) ? stg0 : stg1,
                       weL + ((long)hh * SEQ + k0 + 32) * LAT + n0, 128, 32, LAT, tid);
        tdm_wait_n<1>();
      } else {
        tdm_wait_n<0>();
      }
      __syncthreads();
      convert_transpose(btile, cur, 32, 128, tid);   // [k][128] -> [n=128][k=32]
      convert_a32(abuf, sbuf + k0, SEQ, tid);
      __syncthreads();
      v16bf bf = load_b_nmajor_bf16(btile + (wave * 16) * 32, 32, lane);
      v16bf a0 = load_a_bf16(abuf, 32, lane);
      v16bf a1 = load_a_bf16(abuf + 16 * 32, 32, lane);
      acc0 = wmma_bf16(a0, bf, acc0);
      acc1 = wmma_bf16(a1, bf, acc1);
      __syncthreads();
    }
    int ncol = n0 + wave * 16 + (lane & 15);
    float bb = beL[hh * LAT + ncol];
#pragma unroll
    for (int r = 0; r < 8; ++r) {
      int rr = ((lane >> 4) ? 8 : 0) + r;
      latbuf[rr * LAT + ncol] = fmaxf(acc0[r] + bb, 0.f);
      latbuf[(16 + rr) * LAT + ncol] = fmaxf(acc1[r] + bb, 0.f);
    }
  }
  __syncthreads();

  // 3) rec = sigmoid(lat @ wd + bd), masked -> -inf  (overwrites sbuf)
  for (int sc = 0; sc < SEQ / 128; ++sc) {
    int n0 = sc * 128;
    v8f acc0 = {}, acc1 = {};
    stage_tile_f32(stg0, wdL + (long)hh * LAT * SEQ + n0, 128, 32, SEQ, tid);
    for (int ks = 0; ks < LAT / 32; ++ks) {
      int k0 = ks * 32;
      float* cur = (ks & 1) ? stg1 : stg0;
      if (ks + 1 < LAT / 32) {
        stage_tile_f32((ks & 1) ? stg0 : stg1,
                       wdL + ((long)hh * LAT + k0 + 32) * SEQ + n0, 128, 32, SEQ, tid);
        tdm_wait_n<1>();
      } else {
        tdm_wait_n<0>();
      }
      __syncthreads();
      convert_transpose(btile, cur, 32, 128, tid);
      convert_a32(abuf, latbuf + k0, LAT, tid);
      __syncthreads();
      v16bf bf = load_b_nmajor_bf16(btile + (wave * 16) * 32, 32, lane);
      v16bf a0 = load_a_bf16(abuf, 32, lane);
      v16bf a1 = load_a_bf16(abuf + 16 * 32, 32, lane);
      acc0 = wmma_bf16(a0, bf, acc0);
      acc1 = wmma_bf16(a1, bf, acc1);
      __syncthreads();
    }
    int ncol = n0 + wave * 16 + (lane & 15);
    float bb = bdL[hh * SEQ + ncol];
    bool msk = mask[b * SEQ + ncol] != 0;
#pragma unroll
    for (int r = 0; r < 8; ++r) {
      int rr = ((lane >> 4) ? 8 : 0) + r;
      float v0 = 1.f / (1.f + __expf(-(acc0[r] + bb)));
      float v1 = 1.f / (1.f + __expf(-(acc1[r] + bb)));
      if (msk) { v0 = -__builtin_inff(); v1 = -__builtin_inff(); }
      sbuf[rr * SEQ + ncol] = v0;
      sbuf[(16 + rr) * SEQ + ncol] = v1;
    }
  }
  __syncthreads();

  // 4) row softmax over sbuf + deterministic loss partial
  {
    int row = tid >> 3, seg = tid & 7;     // 8 threads per row
    float* rp = sbuf + row * SEQ + seg * 128;
    float mx = -__builtin_inff();
    for (int i = 0; i < 128; ++i) mx = fmaxf(mx, rp[i]);
    red[row * 8 + seg] = mx;
    __syncthreads();
    if (seg == 0) {
      float m2 = red[row * 8];
      for (int j = 1; j < 8; ++j) m2 = fmaxf(m2, red[row * 8 + j]);
      rowstat[row] = m2;
    }
    __syncthreads();
    float rmax = rowstat[row];
    float ssum = 0.f;
    for (int i = 0; i < 128; ++i) {
      float e = __expf(rp[i] - rmax);
      rp[i] = e;
      ssum += e;
    }
    red[row * 8 + seg] = ssum;
    __syncthreads();
    if (seg == 0) {
      float s2 = 0.f;
      for (int j = 0; j < 8; ++j) s2 += red[row * 8 + j];
      rowstat[row] = s2;
    }
    __syncthreads();
    float inv = 1.f / rowstat[row];
    float nsum = 0.f;
    for (int i = 0; i < 128; ++i) {
      float w = rp[i] * inv;
      rp[i] = w;
      nsum += fabsf(w);
    }
    __syncthreads();
    red[row * 8 + seg] = nsum;
    __syncthreads();
    if (tid == 0) {
      float tl = 0.f;
      for (int j = 0; j < ROWS * 8; ++j) tl += red[j];
      lossSlot[blockIdx.y * gridDim.x + blockIdx.x] = 0.001f * tl;
    }
    __syncthreads();
  }

  // 5) o = w @ V   (32 x 64); wave -> (m0,n0) subtile
  {
    int m0 = (wave >> 2) * 16;
    int n0 = (wave & 3) * 16;
    v8f acc = {};
    stage_tile_f32(stg0, v + qkvb, HDM, 32, EMB, tid);
    for (int ks = 0; ks < SEQ / 32; ++ks) {
      int k0 = ks * 32;
      float* cur = (ks & 1) ? stg1 : stg0;
      if (ks + 1 < SEQ / 32) {
        stage_tile_f32((ks & 1) ? stg0 : stg1, v + qkvb + (long)(k0 + 32) * EMB,
                       HDM, 32, EMB, tid);
        tdm_wait_n<1>();
      } else {
        tdm_wait_n<0>();
      }
      __syncthreads();
      convert_transpose(btile, cur, 32, 64, tid);    // [s][d] -> [n=64][k=32]
      convert_a32(abuf, sbuf + k0, SEQ, tid);
      __syncthreads();
      v16bf bf = load_b_nmajor_bf16(btile + n0 * 32, 32, lane);
      v16bf a = load_a_bf16(abuf + m0 * 32, 32, lane);
      acc = wmma_bf16(a, bf, acc);
      __syncthreads();
    }
    int ncol = n0 + (lane & 15);
#pragma unroll
    for (int r = 0; r < 8; ++r) {
      int rr = m0 + ((lane >> 4) ? 8 : 0) + r;
      o[qkvb + (long)(row0 + rr) * EMB + ncol] = acc[r];
    }
  }
}

// ---------------------------------------------------------------- layernorm(h + delta)
__global__ void ln_residual_kernel(float* __restrict__ h, const float* __restrict__ delta,
                                   const float* __restrict__ g, const float* __restrict__ bb) {
  __shared__ float red[256];
  int row = blockIdx.x, t = threadIdx.x;
  long base = (long)row * EMB;
  float x0 = h[base + t] + delta[base + t];
  float x1 = h[base + t + 256] + delta[base + t + 256];
  red[t] = x0 + x1;
  __syncthreads();
  for (int off = 128; off; off >>= 1) {
    if (t < off) red[t] += red[t + off];
    __syncthreads();
  }
  float mu = red[0] * (1.f / EMB);
  __syncthreads();
  float d0 = x0 - mu, d1 = x1 - mu;
  red[t] = d0 * d0 + d1 * d1;
  __syncthreads();
  for (int off = 128; off; off >>= 1) {
    if (t < off) red[t] += red[t + off];
    __syncthreads();
  }
  float inv = rsqrtf(red[0] * (1.f / EMB) + 1e-5f);
  h[base + t]       = d0 * inv * g[t] + bb[t];
  h[base + t + 256] = d1 * inv * g[t + 256] + bb[t + 256];
}

// ---------------------------------------------------------------- tail kernels
__global__ void copy_kernel(const float* __restrict__ src, float* __restrict__ dst, long n) {
  long i = (long)blockIdx.x * blockDim.x + threadIdx.x;
  if (i < n) dst[i] = src[i];
}

__global__ void loss_reduce_kernel(const float* __restrict__ partial, int n,
                                   float* __restrict__ out) {
  if (blockIdx.x == 0 && threadIdx.x == 0) {
    float s = 0.f;
    for (int i = 0; i < n; ++i) s += partial[i];
    *out = s;
  }
}

// ---------------------------------------------------------------- launch
extern "C" void kernel_launch(void* const* d_in, const int* in_sizes, int n_in,
                              void* d_out, int out_size, void* d_ws, size_t ws_size,
                              hipStream_t stream) {
  const float* tok_emb = (const float*)d_in[0];
  const float* pos_emb = (const float*)d_in[1];
  const float* wq = (const float*)d_in[2];
  const float* bq = (const float*)d_in[3];
  const float* wk = (const float*)d_in[4];
  const float* bk = (const float*)d_in[5];
  const float* wv = (const float*)d_in[6];
  const float* bv = (const float*)d_in[7];
  const float* we = (const float*)d_in[8];
  const float* be = (const float*)d_in[9];
  const float* wd = (const float*)d_in[10];
  const float* bd = (const float*)d_in[11];
  const float* wo = (const float*)d_in[12];
  const float* bo = (const float*)d_in[13];
  const float* ln1_g = (const float*)d_in[14];
  const float* ln1_b = (const float*)d_in[15];
  const float* w1 = (const float*)d_in[16];
  const float* b1 = (const float*)d_in[17];
  const float* w2 = (const float*)d_in[18];
  const float* b2 = (const float*)d_in[19];
  const float* ln2_g = (const float*)d_in[20];
  const float* ln2_b = (const float*)d_in[21];
  const int* x = (const int*)d_in[22];
  const unsigned char* mask = (const unsigned char*)d_in[23];

  const long NTOK = (long)BAT * SEQ;          // 4096
  const long HSZ  = NTOK * EMB;               // 2,097,152
  float* ws   = (float*)d_ws;
  float* h    = ws;
  float* q    = h + HSZ;
  float* k    = q + HSZ;
  float* v    = k + HSZ;
  float* o    = v + HSZ;
  float* proj = o + HSZ;
  float* ff1  = proj + HSZ;
  float* ff2  = ff1 + NTOK * FFN;
  float* lossPartial = ff2 + HSZ;             // LYR * 1024 floats

  const int ATTN_SMEM =
      (int)((ROWS * SEQ + ROWS * LAT + ROWS * 8 + ROWS + 2 * 8192) * sizeof(float) +
            (ROWS * HDM + 128 * HDM + 32 * 32) * sizeof(unsigned short));
  (void)hipFuncSetAttribute((const void*)attention_kernel,
                            hipFuncAttributeMaxDynamicSharedMemorySize, ATTN_SMEM);

  embed_kernel<<<NTOK, 256, 0, stream>>>(tok_emb, pos_emb, x, h);

  for (int l = 0; l < LYR; ++l) {
    const float* wq_l = wq + (long)l * NH * EMB * HDM;
    const float* wk_l = wk + (long)l * NH * EMB * HDM;
    const float* wv_l = wv + (long)l * NH * EMB * HDM;
    const float* bq_l = bq + (long)l * NH * HDM;
    const float* bk_l = bk + (long)l * NH * HDM;
    const float* bv_l = bv + (long)l * NH * HDM;
    const float* we_l = we + (long)l * NH * SEQ * LAT;
    const float* be_l = be + (long)l * NH * LAT;
    const float* wd_l = wd + (long)l * NH * LAT * SEQ;
    const float* bd_l = bd + (long)l * NH * SEQ;
    const float* wo_l = wo + (long)l * EMB * EMB;
    const float* bo_l = bo + (long)l * EMB;
    const float* w1_l = w1 + (long)l * EMB * FFN;
    const float* b1_l = b1 + (long)l * FFN;
    const float* w2_l = w2 + (long)l * FFN * EMB;
    const float* b2_l = b2 + (long)l * EMB;

    dim3 gq(NTOK / 32, EMB / 128);
    gemm_kernel<0, 1><<<gq, 256, 0, stream>>>(h, wq_l, bq_l, q, NTOK, EMB, EMB);
    gemm_kernel<0, 1><<<gq, 256, 0, stream>>>(h, wk_l, bk_l, k, NTOK, EMB, EMB);
    gemm_kernel<0, 1><<<gq, 256, 0, stream>>>(h, wv_l, bv_l, v, NTOK, EMB, EMB);

    dim3 ga(SEQ / ROWS, BAT * NH);
    attention_kernel<<<ga, 256, ATTN_SMEM, stream>>>(
        q, k, v, we_l, wd_l, be_l, bd_l, mask, o, lossPartial + (long)l * 1024);

    gemm_kernel<0, 0><<<gq, 256, 0, stream>>>(o, wo_l, bo_l, proj, NTOK, EMB, EMB);
    ln_residual_kernel<<<NTOK, 256, 0, stream>>>(h, proj,
                                                 ln1_g + (long)l * EMB, ln1_b + (long)l * EMB);

    dim3 gf1(NTOK / 32, FFN / 128);
    gemm_kernel<1, 0><<<gf1, 256, 0, stream>>>(h, w1_l, b1_l, ff1, NTOK, FFN, EMB);
    dim3 gf2(NTOK / 32, EMB / 128);
    gemm_kernel<1, 0><<<gf2, 256, 0, stream>>>(ff1, w2_l, b2_l, ff2, NTOK, EMB, FFN);
    ln_residual_kernel<<<NTOK, 256, 0, stream>>>(h, ff2,
                                                 ln2_g + (long)l * EMB, ln2_b + (long)l * EMB);
  }

  copy_kernel<<<(int)((HSZ + 255) / 256), 256, 0, stream>>>(h, (float*)d_out, HSZ);
  loss_reduce_kernel<<<1, 1, 0, stream>>>(lossPartial, LYR * 1024, (float*)d_out + HSZ);
}
